// ExplicitClusteringModule_57990648431170
// MI455X (gfx1250) — matrix operand order repaired and checked by reading, not compiled, not run on hardware
//
#include <hip/hip_runtime.h>
#include <hip/hip_fp16.h>
#include <hip/hip_bf16.h>

typedef __attribute__((ext_vector_type(16))) _Float16 v16h;
typedef __attribute__((ext_vector_type(8)))  _Float16 v8h;
typedef __attribute__((ext_vector_type(8)))  float    v8f;
typedef __attribute__((ext_vector_type(4)))  float    f4;

#define N_ROWS 131072
#define DIM    1024
#define KCENT  256
#define A_STRIDE 68   // floats per LDS A row (64 data + 4 pad) -> 272B, 16B aligned, conflict-free
#define B_STRIDE 72   // halves per LDS B row (64 data + 8 pad) -> 144B, 16B aligned, conflict-free
static constexpr float EPS = 1e-8f;

// ---------------- kernel 1: normalize centroids -> f16 in workspace ----------
// One wave per centroid row; 512KB f16 copy stays resident in L2 (192MB).
__global__ __launch_bounds__(256)
void centroid_norm_f16_kernel(const float* __restrict__ cent, _Float16* __restrict__ cn)
{
    const int wave = threadIdx.x >> 5;
    const int lane = threadIdx.x & 31;
    const int c = blockIdx.x * 8 + wave;
    const float* src = cent + (size_t)c * DIM;
    float v[32];
    float s = 0.f;
#pragma unroll
    for (int i = 0; i < 32; ++i) {           // coalesced: lane fastest
        float x = src[lane + 32 * i];
        v[i] = x;
        s += x * x;
    }
#pragma unroll
    for (int d = 1; d < 32; d <<= 1) s += __shfl_xor(s, d, 32);
    const float inv = 1.0f / fmaxf(sqrtf(s), EPS);
    _Float16* dst = cn + (size_t)c * DIM;
#pragma unroll
    for (int i = 0; i < 32; ++i) dst[lane + 32 * i] = (_Float16)(v[i] * inv);
}

// Truncate a generic __shared__ pointer to the 32-bit LDS offset (aperture:
// generic LDS addr = {SHARED_BASE[63:32], lds_offset[31:0]}).
__device__ __forceinline__ unsigned lds_off(const void* p) {
    return (unsigned)(size_t)p;
}

// Issue 8 async B128 transfers (128B) memory->LDS; both sides contiguous,
// IOFFSET applies to the global AND the LDS address per the CDNA5 ISA.
__device__ __forceinline__ void async_copy128(unsigned lds, const char* g) {
#pragma unroll
    for (int i = 0; i < 8; ++i) {
        asm volatile("global_load_async_to_lds_b128 %0, %1, off offset:%2"
                     :: "v"(lds), "v"(g), "i"(i * 16) : "memory");
    }
}

// ---------------- kernel 2: fused GEMM (WMMA f16) + row-norm + softmax -------
// Block: 256 thr = 8 waves. Block tile: 128 rows x 256 cols. Wave: 16 rows x 256 cols
// -> 16 accumulator tiles (16x16 f32 = 8 VGPR each). d streamed in chunks of 64,
// double-buffered in LDS via async global->LDS loads (ASYNCcnt pipelining).
__global__ __launch_bounds__(256)
void cosine_softmax_kernel(const float* __restrict__ batch,
                           const _Float16* __restrict__ cn,
                           float* __restrict__ out)
{
    __shared__ __align__(32) float    lA[2][128 * A_STRIDE];  // 2 x 34816 B
    __shared__ __align__(32) _Float16 lB[2][256 * B_STRIDE];  // 2 x 36864 B

    const int tid  = threadIdx.x;
    const int wave = tid >> 5;
    const int lane = tid & 31;
    const int hi   = lane >> 4;   // half-wave select (ISA fragment layouts)
    const int lm   = lane & 15;
    const int rowblk = blockIdx.x * 128;
    const int rw     = wave * 16; // this wave's row strip within the block

    v8f acc[16];
    v8f zero = {};
#pragma unroll
    for (int t = 0; t < 16; ++t) acc[t] = zero;
    float sumsq = 0.f;            // per-lane partial of ||x||^2 for row (rw+lm)

    const int ar = tid >> 1, apart = tid & 1;   // A-slice loader mapping

    // Per-thread staging bases (A: 128 rows x 64 f32; B: 256 cols x 64 f16)
    const char* gA0 = (const char*)(batch + (size_t)(rowblk + ar) * DIM + apart * 32);
    const char* gB0 = (const char*)(cn + (size_t)tid * DIM);
    const unsigned lA0 = lds_off(&lA[0][ar * A_STRIDE + apart * 32]);
    const unsigned lA1 = lds_off(&lA[1][ar * A_STRIDE + apart * 32]);
    const unsigned lB0 = lds_off(&lB[0][tid * B_STRIDE]);
    const unsigned lB1 = lds_off(&lB[1][tid * B_STRIDE]);

    // Prologue: stage chunk 0 into buffer 0.
    async_copy128(lA0, gA0);
    async_copy128(lB0, gB0);

    constexpr int NCHUNK = DIM / 64;   // 16
    for (int c = 0; c < NCHUNK; ++c) {
        asm volatile("s_wait_asynccnt 0x0" ::: "memory");  // chunk c landed in LDS
        __syncthreads();  // visible to all waves; frees the other buffer for reuse

        if (c + 1 < NCHUNK) {   // stage chunk c+1 into the other buffer
            const int nb = (c + 1) & 1;
            async_copy128(nb ? lA1 : lA0, gA0 + (size_t)(c + 1) * 64 * 4);
            async_copy128(nb ? lB1 : lB0, gB0 + (size_t)(c + 1) * 64 * 2);
        }

        const int buf = c & 1;
        const float*    lAb = lA[buf];
        const _Float16* lBb = lB[buf];

#pragma unroll
        for (int ks = 0; ks < 2; ++ks) {
            // A fragment per ISA 16-bit A 16x32 layout:
            //   j<8 : k = j + 8*hi ; j>=8 : k = 16 + (j-8) + 8*hi
            const f4* ap4 = (const f4*)(lAb + (rw + lm) * A_STRIDE + ks * 32 + 8 * hi);
            f4 a00 = ap4[0], a01 = ap4[1];   // d = base + 0..7
            f4 a10 = ap4[4], a11 = ap4[5];   // d = base + 16..23
            sumsq += a00.x*a00.x + a00.y*a00.y + a00.z*a00.z + a00.w*a00.w
                   + a01.x*a01.x + a01.y*a01.y + a01.z*a01.z + a01.w*a01.w
                   + a10.x*a10.x + a10.y*a10.y + a10.z*a10.z + a10.w*a10.w
                   + a11.x*a11.x + a11.y*a11.y + a11.z*a11.z + a11.w*a11.w;
            v16h af;
            af[0]=(_Float16)a00.x; af[1]=(_Float16)a00.y; af[2]=(_Float16)a00.z; af[3]=(_Float16)a00.w;
            af[4]=(_Float16)a01.x; af[5]=(_Float16)a01.y; af[6]=(_Float16)a01.z; af[7]=(_Float16)a01.w;
            af[8]=(_Float16)a10.x; af[9]=(_Float16)a10.y; af[10]=(_Float16)a10.z; af[11]=(_Float16)a10.w;
            af[12]=(_Float16)a11.x; af[13]=(_Float16)a11.y; af[14]=(_Float16)a11.z; af[15]=(_Float16)a11.w;

#pragma unroll
            for (int t = 0; t < 16; ++t) {
                // B fragment per ISA 16-bit B 32x16 layout: lane col = lm, j<8 : k = j + 8*hi ; j>=8 : k = 16 + (j-8) + 8*hi
                const v8h* bp = (const v8h*)(lBb + (16 * t + lm) * B_STRIDE + ks * 32 + hi * 8);
                v8h b0 = bp[0], b1 = bp[2];
                v16h bf = __builtin_shufflevector(b0, b1, 0,1,2,3,4,5,6,7,8,9,10,11,12,13,14,15);
                acc[t] = __builtin_amdgcn_wmma_f32_16x16x32_f16(
                    false, af, false, bf, (short)0, acc[t], false, false);
            }
        }
    }

    // ---- per-row 1/||x|| : lane pair (lm, lm+16) hold disjoint d-subsets ----
    sumsq += __shfl_xor(sumsq, 16, 32);
    const float invn = 1.0f / fmaxf(sqrtf(sumsq), EPS);   // valid for row rw+lm on all lanes
    float inr[8];
#pragma unroll
    for (int r = 0; r < 8; ++r) inr[r] = __shfl(invn, r + 8 * hi, 32); // norm of this lane's acc row r

    // ---- softmax over K=256 per row (row r+8*hi lives in the lane's half) ----
#pragma unroll
    for (int r = 0; r < 8; ++r) {
        float m = -3.402823466e38f;
#pragma unroll
        for (int t = 0; t < 16; ++t) { acc[t][r] *= inr[r]; m = fmaxf(m, acc[t][r]); }
#pragma unroll
        for (int dd = 1; dd < 16; dd <<= 1) m = fmaxf(m, __shfl_xor(m, dd, 32)); // stays in half
        float s = 0.f;
#pragma unroll
        for (int t = 0; t < 16; ++t) { float e = __expf(acc[t][r] - m); acc[t][r] = e; s += e; }
#pragma unroll
        for (int dd = 1; dd < 16; dd <<= 1) s += __shfl_xor(s, dd, 32);
        const float rs = 1.0f / s;
        float* orow = out + (size_t)(rowblk + rw + r + 8 * hi) * KCENT;
#pragma unroll
        for (int t = 0; t < 16; ++t) orow[16 * t + lm] = acc[t][r] * rs;  // 64B segments/half
    }
}

extern "C" void kernel_launch(void* const* d_in, const int* in_sizes, int n_in,
                              void* d_out, int out_size, void* d_ws, size_t ws_size,
                              hipStream_t stream) {
    const float* batch = (const float*)d_in[0];
    const float* cent  = (const float*)d_in[1];
    float* out = (float*)d_out;
    _Float16* cn = (_Float16*)d_ws;   // 256*1024*2 = 512 KB of workspace

    centroid_norm_f16_kernel<<<KCENT / 8, 256, 0, stream>>>(cent, cn);
    cosine_softmax_kernel<<<N_ROWS / 128, 256, 0, stream>>>(batch, cn, out);
}
